// SeqLabelModel_34136400068873
// MI455X (gfx1250) — compile-verified, matrix operand order checked by reading, not run on hardware
//
#include <hip/hip_runtime.h>
#include <hip/hip_bf16.h>
#include <math.h>

typedef __bf16 bf16_t;
typedef __attribute__((ext_vector_type(16))) __bf16 v16bf;
typedef __attribute__((ext_vector_type(8)))  __bf16 v8bf;
typedef __attribute__((ext_vector_type(8)))  float  v8f;
typedef __attribute__((ext_vector_type(4)))  unsigned int v4u;
typedef __attribute__((ext_vector_type(8)))  int v8i;
typedef __attribute__((ext_vector_type(4)))  int v4i;

#define S_LEN 512
#define BATCH 32
#define EDIM  256
#define HDIM  256
#define GDIM  1024   /* 4*H */
#define KTAG  4
#define NEGV  -10000.0f

// ---- WMMA helpers -----------------------------------------------------------
// 16-bit A fragment (16x32 MxK) / B fragment (32x16 KxN) element mapping:
// lane L, bf16 index j -> k = (j<8?0:16) + (L>=16?8:0) + (j&7)
// => two contiguous 8-element (16-byte) chunks at offsets {hi8, 16+hi8}.
__device__ __forceinline__ v16bf load_frag16(const bf16_t* p, int hi8) {
  const v8bf lo = *reinterpret_cast<const v8bf*>(p + hi8);
  const v8bf hi = *reinterpret_cast<const v8bf*>(p + 16 + hi8);
  v16bf r;
#pragma unroll
  for (int j = 0; j < 8; ++j) { r[j] = lo[j]; r[j + 8] = hi[j]; }
  return r;
}

__device__ __forceinline__ v8f wmma_bf16(v16bf a, v16bf b, v8f c) {
  return __builtin_amdgcn_wmma_f32_16x16x32_bf16(false, a, false, b,
                                                 (short)0, c, false, false);
}

__device__ __forceinline__ float sigf(float x) {
  return 1.0f / (1.0f + __expf(-x));
}

// Flat shared pointer -> LDS byte address (keeps the low 32 bits of the address)
__device__ __forceinline__ unsigned lds_addr_of(const void* p) {
  return (unsigned)(unsigned long long)(uintptr_t)p;
}

// TDM: async copy of `nelem` contiguous bf16 elements from global -> LDS.
// Builds a 1-D Tensor DMA Descriptor per CDNA5 ISA.
__device__ __forceinline__ void tdm_copy_1d_bf16(unsigned lds_byte_addr,
                                                 const void* gptr,
                                                 unsigned nelem) {
  unsigned long long ga = (unsigned long long)(uintptr_t)gptr;
  v4u g0;
  g0[0] = 1u;                                        // count=1, user mode
  g0[1] = lds_byte_addr;                             // lds_addr
  g0[2] = (unsigned)(ga & 0xFFFFFFFFu);              // global_addr[31:0]
  g0[3] = (unsigned)((ga >> 32) & 0x01FFFFFFu)       // global_addr[56:32]
        | (2u << 30);                                // type = 2 ("image")
  v8i g1;
  g1[0] = (int)(1u << 16);                           // wg_mask=0, data_size=1 (2B)
  g1[1] = (int)((nelem & 0xFFFFu) << 16);            // tensor_dim0[15:0]
  g1[2] = (int)((nelem >> 16) & 0xFFFFu);            // tensor_dim0[31:16], tensor_dim1=0
  g1[3] = (int)((nelem & 0xFFFFu) << 16);            // tile_dim0 = nelem
  g1[4] = 0;                                         // tile_dim1=0, tile_dim2=0
  g1[5] = (int)nelem;                                // tensor_dim0_stride[31:0]
  g1[6] = 0;
  g1[7] = 0;
  v4i gz = {0, 0, 0, 0};
#if defined(__clang_major__) && (__clang_major__ >= 23)
  v8i gz8 = {0, 0, 0, 0, 0, 0, 0, 0};
  __builtin_amdgcn_tensor_load_to_lds(g0, g1, gz, gz, gz8, 0);
#else
  __builtin_amdgcn_tensor_load_to_lds(g0, g1, gz, gz, 0);
#endif
}

// ---- Kernel 1: embedding gather -> bf16 x[S][B][E] --------------------------
__global__ void k_embed(const int* __restrict__ sent,
                        const float* __restrict__ emb,
                        bf16_t* __restrict__ x) {
  size_t i  = (size_t)blockIdx.x * blockDim.x + threadIdx.x;
  size_t e8 = i * 8;
  if (e8 >= (size_t)S_LEN * BATCH * EDIM) return;
  int e  = (int)(e8 % EDIM);
  int sb = (int)(e8 / EDIM);
  int b  = sb % BATCH;
  int s  = sb / BATCH;
  int tok = sent[b * S_LEN + s];            // sentence is [B,S]
  const float* src = emb + (size_t)tok * EDIM + e;
  bf16_t* dst = x + e8;
#pragma unroll
  for (int q = 0; q < 8; ++q) dst[q] = (bf16_t)src[q];
}

// ---- Kernel 2: weight conversion + combined biases --------------------------
__global__ void k_prep(const float* wih_f, const float* whh_f,
                       const float* bih_f, const float* bhh_f,
                       const float* wih_b, const float* whh_b,
                       const float* bih_b, const float* bhh_b,
                       bf16_t* wih_f_h, bf16_t* whh_f_h,
                       bf16_t* wih_b_h, bf16_t* whh_b_h,
                       float* bias_f, float* bias_b) {
  int i = blockIdx.x * blockDim.x + threadIdx.x;
  const int n = GDIM * EDIM;                // 262144 (E==H so same count)
  if (i < n) {
    wih_f_h[i] = (bf16_t)wih_f[i];
    whh_f_h[i] = (bf16_t)whh_f[i];
    wih_b_h[i] = (bf16_t)wih_b[i];
    whh_b_h[i] = (bf16_t)whh_b[i];
  }
  if (i < GDIM) {
    bias_f[i] = bih_f[i] + bhh_f[i];
    bias_b[i] = bih_b[i] + bhh_b[i];
  }
}

// ---- Kernel 3: xg = x * wih^T + bias  (big parallel WMMA GEMM) -------------
// grid (R/32, G/64), block 256 (8 waves). Wave w: m-tile = w&1, n-tile = w>>1.
__global__ void k_xg_gemm(const bf16_t* __restrict__ x,
                          const bf16_t* __restrict__ wih,
                          const float*  __restrict__ bias,
                          bf16_t* __restrict__ xg) {
  const int lane = threadIdx.x & 31;
  const int wave = threadIdx.x >> 5;
  const int mt   = wave & 1;
  const int ntp  = wave >> 1;                       // 0..3
  const int hi8  = (lane >= 16) ? 8 : 0;
  const int rowA = blockIdx.x * 32 + mt * 16 + (lane & 15);
  const int colN = blockIdx.y * 64 + ntp * 16 + (lane & 15);
  const bf16_t* arow = x   + (size_t)rowA * EDIM;
  const bf16_t* brow = wih + (size_t)colN * EDIM;
  v8f acc = {};
#pragma unroll
  for (int kt = 0; kt < EDIM / 32; ++kt) {
    v16bf a = load_frag16(arow + kt * 32, hi8);
    v16bf b = load_frag16(brow + kt * 32, hi8);
    acc = wmma_bf16(a, b, acc);
  }
  const int mbase = blockIdx.x * 32 + mt * 16 + ((lane >= 16) ? 8 : 0);
  const float bc = bias[colN];
#pragma unroll
  for (int r = 0; r < 8; ++r) {
    xg[(size_t)(mbase + r) * GDIM + colN] = (bf16_t)(acc[r] + bc);
  }
}

// ---- Kernel 4: sequential LSTM recurrence (one WGP per direction) -----------
// block = 1024 threads (32 waves); grid = 2 (dir 0 fwd, dir 1 bwd).
// LDS: xg[32][1024] bf16 (64 KB, TDM-filled) + h[32][256] bf16 (16 KB)
//      + gates[32][1024] f32 (128 KB) = 208 KB dynamic (WGP has 320 KB).
__global__ void k_lstm(const bf16_t* __restrict__ xg_f,
                       const bf16_t* __restrict__ xg_b,
                       const bf16_t* __restrict__ whh_f,
                       const bf16_t* __restrict__ whh_b,
                       const float*  __restrict__ h0,
                       const float*  __restrict__ c0,
                       bf16_t* __restrict__ hs) {
  extern __shared__ char smem[];
  bf16_t* xg_lds = (bf16_t*)smem;                                  // [32][1024]
  bf16_t* h_lds  = (bf16_t*)(smem + (size_t)BATCH * GDIM * sizeof(bf16_t));
  float*  g_lds  = (float*)(smem + (size_t)BATCH * GDIM * sizeof(bf16_t)
                                 + (size_t)BATCH * HDIM * sizeof(bf16_t));

  const int dir  = blockIdx.x;
  const bf16_t* xg  = dir ? xg_b  : xg_f;
  const bf16_t* whh = dir ? whh_b : whh_f;

  const int tid  = threadIdx.x;
  const int lane = tid & 31;
  const int wave = tid >> 5;

  // elementwise-phase ownership: thread handles 8 consecutive (b,j) elems
  const int eb = tid / 32;                 // batch row
  const int jb = (tid * 8) & 255;          // hidden-col base

  // matmul-phase ownership: wave -> (m-tile, 4 n-tiles)
  const int mt     = wave & 1;
  const int ntbase = (wave >> 1) * 4;      // 0,4,...,60
  const int hi8    = (lane >= 16) ? 8 : 0;
  const int mrowA  = mt * 16 + (lane & 15);

  const unsigned xg_lds_base = lds_addr_of(xg_lds);

  float c_reg[8];
  {
    const float* h0d = h0 + (size_t)dir * BATCH * HDIM;
    const float* c0d = c0 + (size_t)dir * BATCH * HDIM;
#pragma unroll
    for (int q = 0; q < 8; ++q) {
      int e = tid * 8 + q;
      h_lds[e] = (bf16_t)h0d[e];
      c_reg[q] = c0d[e];
    }
  }
  __syncthreads();

#pragma unroll 1
  for (int t = 0; t < S_LEN; ++t) {
    const int ts = dir ? (S_LEN - 1 - t) : t;

    // ---- TDM: pull this step's 64 KB xg slice into LDS; overlaps the WMMAs.
    if (wave == 0) {
      tdm_copy_1d_bf16(xg_lds_base, xg + (size_t)ts * BATCH * GDIM,
                       (unsigned)(BATCH * GDIM));
    }

    // ---- gates = h @ whh^T : 2x64 tiles of 16x16, 4 per wave --------------
    v8f acc0 = {}, acc1 = {}, acc2 = {}, acc3 = {};
    const bf16_t* arow = h_lds + (size_t)mrowA * HDIM;
    const bf16_t* bcol = whh + (size_t)(ntbase * 16 + (lane & 15)) * HDIM;
#pragma unroll
    for (int kt = 0; kt < HDIM / 32; ++kt) {
      v16bf a  = load_frag16(arow + kt * 32, hi8);
      v16bf b0 = load_frag16(bcol + (size_t)0 * 16 * HDIM + kt * 32, hi8);
      v16bf b1 = load_frag16(bcol + (size_t)1 * 16 * HDIM + kt * 32, hi8);
      v16bf b2 = load_frag16(bcol + (size_t)2 * 16 * HDIM + kt * 32, hi8);
      v16bf b3 = load_frag16(bcol + (size_t)3 * 16 * HDIM + kt * 32, hi8);
      acc0 = wmma_bf16(a, b0, acc0);
      acc1 = wmma_bf16(a, b1, acc1);
      acc2 = wmma_bf16(a, b2, acc2);
      acc3 = wmma_bf16(a, b3, acc3);
    }
    {
      const int mb = mt * 16 + ((lane >= 16) ? 8 : 0);
      const int nc = (lane & 15);
#pragma unroll
      for (int r = 0; r < 8; ++r) {
        g_lds[(mb + r) * GDIM + (ntbase + 0) * 16 + nc] = acc0[r];
        g_lds[(mb + r) * GDIM + (ntbase + 1) * 16 + nc] = acc1[r];
        g_lds[(mb + r) * GDIM + (ntbase + 2) * 16 + nc] = acc2[r];
        g_lds[(mb + r) * GDIM + (ntbase + 3) * 16 + nc] = acc3[r];
      }
    }
    // Issuing wave drains TENSORcnt; the barrier publishes the LDS tile.
    if (wave == 0) __builtin_amdgcn_s_wait_tensorcnt(0);
    __syncthreads();

    // ---- elementwise cell update (xg now in LDS) --------------------------
    {
      const bf16_t* xr = xg_lds + (size_t)eb * GDIM;
      const float*  gr = g_lds  + (size_t)eb * GDIM;
      bf16_t* hrow = h_lds + (size_t)eb * HDIM;
      bf16_t* hsr  = hs + ((size_t)ts * BATCH + eb) * (2 * HDIM) + dir * HDIM;
#pragma unroll
      for (int q = 0; q < 8; ++q) {
        int j = jb + q;
        float gi = gr[j]       + (float)xr[j];
        float gf = gr[256 + j] + (float)xr[256 + j];
        float gg = gr[512 + j] + (float)xr[512 + j];
        float go = gr[768 + j] + (float)xr[768 + j];
        float cn = sigf(gf) * c_reg[q] + sigf(gi) * tanhf(gg);
        c_reg[q] = cn;
        float hn = sigf(go) * tanhf(cn);
        hrow[j] = (bf16_t)hn;
        hsr[j]  = (bf16_t)hn;
      }
    }
    __syncthreads();   // xg_lds/h_lds consumed before next step's TDM/WMMA
  }
}

// ---- Kernel 5: feats = hs * w_out^T + b_out  (K=4, VALU) --------------------
__global__ void k_proj(const bf16_t* __restrict__ hs,
                       const float*  __restrict__ w_out,
                       const float*  __restrict__ b_out,
                       float* __restrict__ feats) {
  __shared__ float w_lds[KTAG * 2 * HDIM];   // 8 KB
  for (int i = threadIdx.x; i < KTAG * 2 * HDIM; i += blockDim.x)
    w_lds[i] = w_out[i];
  __syncthreads();

  int i = blockIdx.x * blockDim.x + threadIdx.x;
  if (i >= S_LEN * BATCH) return;
  const bf16_t* hr = hs + (size_t)i * (2 * HDIM);
  float acc[KTAG] = {b_out[0], b_out[1], b_out[2], b_out[3]};
#pragma unroll 4
  for (int j = 0; j < 2 * HDIM; ++j) {
    float hv = (float)hr[j];
#pragma unroll
    for (int k = 0; k < KTAG; ++k) acc[k] += hv * w_lds[k * (2 * HDIM) + j];
  }
#pragma unroll
  for (int k = 0; k < KTAG; ++k) feats[(size_t)i * KTAG + k] = acc[k];
}

// ---- Kernel 6: Viterbi decode (one lane per batch element) ------------------
__global__ void k_viterbi(const float* __restrict__ feats,
                          const float* __restrict__ trans,
                          unsigned int* __restrict__ bp,
                          float* __restrict__ out) {
  int b = threadIdx.x;
  if (b >= BATCH) return;
  float T[KTAG][KTAG];
#pragma unroll
  for (int nx = 0; nx < KTAG; ++nx)
#pragma unroll
    for (int pv = 0; pv < KTAG; ++pv) T[nx][pv] = trans[nx * KTAG + pv];

  float fv[KTAG];
#pragma unroll
  for (int k = 0; k < KTAG; ++k) fv[k] = NEGV;
  fv[2] = 0.0f;                               // START = 2

#pragma unroll 1
  for (int s = 0; s < S_LEN; ++s) {
    const float* fr = feats + ((size_t)s * BATCH + b) * KTAG;
    float nf[KTAG];
    unsigned int pk = 0;
#pragma unroll
    for (int nx = 0; nx < KTAG; ++nx) {
      float best = fv[0] + T[nx][0];
      int bi = 0;
#pragma unroll
      for (int pv = 1; pv < KTAG; ++pv) {
        float v = fv[pv] + T[nx][pv];
        if (v > best) { best = v; bi = pv; }
      }
      nf[nx] = best + fr[nx];
      pk |= ((unsigned)bi) << (8 * nx);
    }
    bp[s * BATCH + b] = pk;
#pragma unroll
    for (int k = 0; k < KTAG; ++k) fv[k] = nf[k];
  }

  // terminal = fv + transitions[STOP]; STOP = 3
  float best = fv[0] + T[3][0];
  int last = 0;
#pragma unroll
  for (int k = 1; k < KTAG; ++k) {
    float v = fv[k] + T[3][k];
    if (v > best) { best = v; last = k; }
  }
  out[b] = best;                               // path_score
  float* path = out + BATCH + (size_t)b * S_LEN;
  path[S_LEN - 1] = (float)last;
  int tag = last;
#pragma unroll 1
  for (int s = S_LEN - 1; s >= 1; --s) {
    tag = (int)((bp[s * BATCH + b] >> (8 * tag)) & 0xFFu);
    path[s - 1] = (float)tag;
  }
}

// ---- Host launcher ----------------------------------------------------------
extern "C" void kernel_launch(void* const* d_in, const int* in_sizes, int n_in,
                              void* d_out, int out_size, void* d_ws, size_t ws_size,
                              hipStream_t stream) {
  const int*   sent  = (const int*)  d_in[0];
  const float* h0    = (const float*)d_in[1];
  const float* c0    = (const float*)d_in[2];
  const float* emb   = (const float*)d_in[3];
  const float* wih_f = (const float*)d_in[4];
  const float* whh_f = (const float*)d_in[5];
  const float* bih_f = (const float*)d_in[6];
  const float* bhh_f = (const float*)d_in[7];
  const float* wih_b = (const float*)d_in[8];
  const float* whh_b = (const float*)d_in[9];
  const float* bih_b = (const float*)d_in[10];
  const float* bhh_b = (const float*)d_in[11];
  const float* w_out = (const float*)d_in[12];
  const float* b_out = (const float*)d_in[13];
  const float* trans = (const float*)d_in[14];

  char* ws = (char*)d_ws;
  size_t off = 0;
  auto alloc = [&](size_t bytes) -> void* {
    void* p = ws + off;
    off += (bytes + 255) & ~(size_t)255;
    return p;
  };

  bf16_t* x        = (bf16_t*)alloc((size_t)S_LEN * BATCH * EDIM * sizeof(bf16_t));
  bf16_t* wih_f_h  = (bf16_t*)alloc((size_t)GDIM * EDIM * sizeof(bf16_t));
  bf16_t* whh_f_h  = (bf16_t*)alloc((size_t)GDIM * HDIM * sizeof(bf16_t));
  bf16_t* wih_b_h  = (bf16_t*)alloc((size_t)GDIM * EDIM * sizeof(bf16_t));
  bf16_t* whh_b_h  = (bf16_t*)alloc((size_t)GDIM * HDIM * sizeof(bf16_t));
  float*  bias_f   = (float*) alloc((size_t)GDIM * sizeof(float));
  float*  bias_b   = (float*) alloc((size_t)GDIM * sizeof(float));
  bf16_t* xg_f     = (bf16_t*)alloc((size_t)S_LEN * BATCH * GDIM * sizeof(bf16_t));
  bf16_t* xg_b     = (bf16_t*)alloc((size_t)S_LEN * BATCH * GDIM * sizeof(bf16_t));
  bf16_t* hs       = (bf16_t*)alloc((size_t)S_LEN * BATCH * 2 * HDIM * sizeof(bf16_t));
  float*  feats    = (float*) alloc((size_t)S_LEN * BATCH * KTAG * sizeof(float));
  unsigned int* bp = (unsigned int*)alloc((size_t)S_LEN * BATCH * sizeof(unsigned int));

  // 1) embedding gather
  {
    size_t total = (size_t)S_LEN * BATCH * EDIM / 8;
    k_embed<<<(unsigned)((total + 255) / 256), 256, 0, stream>>>(sent, emb, x);
  }
  // 2) weight prep
  k_prep<<<(GDIM * EDIM + 255) / 256, 256, 0, stream>>>(
      wih_f, whh_f, bih_f, bhh_f, wih_b, whh_b, bih_b, bhh_b,
      wih_f_h, whh_f_h, wih_b_h, whh_b_h, bias_f, bias_b);
  // 3) input-gate GEMMs (both directions)
  {
    dim3 grid(S_LEN * BATCH / 32, GDIM / 64);
    k_xg_gemm<<<grid, 256, 0, stream>>>(x, wih_f_h, bias_f, xg_f);
    k_xg_gemm<<<grid, 256, 0, stream>>>(x, wih_b_h, bias_b, xg_b);
  }
  // 4) sequential recurrence: 208 KB dynamic LDS per workgroup (WGP has 320 KB)
  {
    const size_t smem = (size_t)BATCH * GDIM * sizeof(bf16_t)      // xg tile
                      + (size_t)BATCH * HDIM * sizeof(bf16_t)      // h
                      + (size_t)BATCH * GDIM * sizeof(float);      // gates
    hipFuncSetAttribute(reinterpret_cast<const void*>(k_lstm),
                        hipFuncAttributeMaxDynamicSharedMemorySize, (int)smem);
    k_lstm<<<2, 1024, smem, stream>>>(xg_f, xg_b, whh_f_h, whh_b_h, h0, c0, hs);
  }
  // 5) CRF emission features
  k_proj<<<(S_LEN * BATCH) / 256, 256, 0, stream>>>(hs, w_out, b_out, feats);
  // 6) Viterbi decode + backtrace
  k_viterbi<<<1, 32, 0, stream>>>(feats, trans, bp, (float*)d_out);
}
